// PointNet2Sem_seg_29076928594036
// MI455X (gfx1250) — matrix operand order repaired and checked by reading, not compiled, not run on hardware
//
#include <hip/hip_runtime.h>
#include <hip/hip_bf16.h>
#include <math.h>

typedef _Float16 v16h __attribute__((ext_vector_type(16)));
typedef _Float16 v8h  __attribute__((ext_vector_type(8)));
typedef float    v8f  __attribute__((ext_vector_type(8)));

#define BATCH 8

// ---------------------------------------------------------------- extract xyz
__global__ void extract_xyz_kernel(const float* __restrict__ x, float* __restrict__ xyz, int total) {
    int id = blockIdx.x * blockDim.x + threadIdx.x;
    if (id >= total) return;
    int pi = id / 3, c = id % 3;
    xyz[id] = x[(size_t)pi * 9 + c];
}

// ---------------------------------------------------------------- per-layer weight prep
// Convert W (Cout,Ktrue) f32 -> Wh (Cout,Kp) f16 zero-padded, and fold bias+BN into
// per-channel (scale, offset):  y = acc*scale + offset  (then ReLU).
__global__ void prep_layer_kernel(const float* __restrict__ W, const float* __restrict__ bias,
                                  const float* __restrict__ gamma, const float* __restrict__ beta,
                                  const float* __restrict__ mean, const float* __restrict__ var,
                                  _Float16* __restrict__ Wh, float2* __restrict__ so,
                                  int Ktrue, int Kp, int Cout) {
    int id = blockIdx.x * blockDim.x + threadIdx.x;
    if (id >= Cout * Kp) return;
    int o = id / Kp, k = id % Kp;
    Wh[id] = (k < Ktrue) ? (_Float16)W[(size_t)o * Ktrue + k] : (_Float16)0.f;
    if (k == 0) {
        float s = gamma[o] * rsqrtf(var[o] + 1e-5f);
        so[o] = make_float2(s, (bias[o] - mean[o]) * s + beta[o]);
    }
}

// ---------------------------------------------------------------- farthest point sampling
__global__ void fps_kernel(const float* __restrict__ xyz, float* __restrict__ newxyz, int n, int np) {
    __shared__ float dist[4096];
    __shared__ float rv[256];
    __shared__ int   ri[256];
    __shared__ int   sfar;
    int b = blockIdx.x, t = threadIdx.x;
    const float* px = xyz + (size_t)b * n * 3;
    for (int i = t; i < n; i += 256) dist[i] = 1e10f;
    if (t == 0) sfar = 0;
    __syncthreads();
    for (int s = 0; s < np; ++s) {
        int far = sfar;
        float cx = px[far*3+0], cy = px[far*3+1], cz = px[far*3+2];
        if (t == 0) { float* o = newxyz + ((size_t)b*np + s)*3; o[0]=cx; o[1]=cy; o[2]=cz; }
        float bv = -1.f; int bi = 0;
        for (int i = t; i < n; i += 256) {
            float dx = px[i*3]-cx, dy = px[i*3+1]-cy, dz = px[i*3+2]-cz;
            float d  = dx*dx + dy*dy + dz*dz;
            float nd = fminf(dist[i], d);
            dist[i] = nd;
            if (nd > bv) { bv = nd; bi = i; }   // strict > keeps first index on ties
        }
        rv[t] = bv; ri[t] = bi;
        __syncthreads();
        for (int o = 128; o > 0; o >>= 1) {
            if (t < o) {
                if (rv[t+o] > rv[t] || (rv[t+o] == rv[t] && ri[t+o] < ri[t])) {
                    rv[t] = rv[t+o]; ri[t] = ri[t+o];
                }
            }
            __syncthreads();
        }
        if (t == 0) sfar = ri[0];
        __syncthreads();
    }
}

// ---------------------------------------------------------------- ball query + group (wave32 ballot)
__global__ void ball_group_kernel(const float* __restrict__ xyz, const float* __restrict__ newxyz,
                                  const float* __restrict__ feat32, const _Float16* __restrict__ feat16,
                                  int cf, _Float16* __restrict__ G, int n, int np, float r2, int Kp) {
    __shared__ int sIdx[4][32];
    int lane = threadIdx.x & 31;
    int w    = threadIdx.x >> 5;
    int gc   = blockIdx.x * 4 + w;       // global center id  (B*np is always a multiple of 4)
    int b    = gc / np;
    const float* ctr = newxyz + (size_t)gc * 3;
    float cx = ctr[0], cy = ctr[1], cz = ctr[2];
    const float* px = xyz + (size_t)b * n * 3;
    int cnt = 0;
    int nchunk = n >> 5;
    for (int c = 0; c < nchunk && cnt < 32; ++c) {
        int p = (c << 5) + lane;
        float dx = px[p*3]-cx, dy = px[p*3+1]-cy, dz = px[p*3+2]-cz;
        bool pred = (dx*dx + dy*dy + dz*dz) <= r2;
        unsigned mask = __builtin_amdgcn_ballot_w32(pred);
        int pos = cnt + __popc(mask & ((1u << lane) - 1u));
        if (pred && pos < 32) sIdx[w][pos] = p;
        cnt += __popc(mask);
    }
    __syncthreads();
    int m = cnt < 32 ? cnt : 32;         // center itself always in radius -> m >= 1
    int idx = (lane < m) ? sIdx[w][lane] : sIdx[w][0];
    _Float16* row = G + ((size_t)gc * 32 + lane) * Kp;
    row[0] = (_Float16)(px[idx*3+0] - cx);
    row[1] = (_Float16)(px[idx*3+1] - cy);
    row[2] = (_Float16)(px[idx*3+2] - cz);
    if (feat32) {
        const float* f = feat32 + ((size_t)b * n + idx) * cf;
        for (int c2 = 0; c2 < cf; ++c2) row[3 + c2] = (_Float16)f[c2];
    } else {
        const _Float16* f = feat16 + ((size_t)b * n + idx) * cf;
        for (int c2 = 0; c2 < cf; ++c2) row[3 + c2] = f[c2];
    }
    for (int c2 = 3 + cf; c2 < Kp; ++c2) row[c2] = (_Float16)0.f;
}

// ---------------------------------------------------------------- WMMA GEMM + fused BN + ReLU
// One 16(M)x32(N) tile per wave: A fragment loaded once, two WMMAs per K-step.
// X : (M, Kp)   f16 row-major (zero-padded), Kp multiple of 32.
// Wh: (Cout,Kp) f16 row-major (pre-converted, pre-padded)  -> B loads are contiguous 16B.
// so: per-channel (scale, offset) with bias+BN folded in.
// grid = (M/64, Cout/32), block = 128 (4 waves). No EXEC divergence anywhere.
__global__ void gemm_bn_relu_kernel(const _Float16* __restrict__ X, int Kp,
                                    const _Float16* __restrict__ Wh,
                                    const float2* __restrict__ so,
                                    _Float16* __restrict__ Y, int Cout) {
    int lane = threadIdx.x & 31;
    int wv   = threadIdx.x >> 5;
    int m0   = (blockIdx.x * 4 + wv) * 16;
    int n0   = blockIdx.y * 32;
    int r16  = lane & 15;
    int hi   = lane >> 4;
    const _Float16* xr  = X  + (size_t)(m0 + r16) * Kp;       // A: row = lane&15
    const _Float16* wr0 = Wh + (size_t)(n0 + r16) * Kp;       // B tile 0: col = n0 + lane&15
    const _Float16* wr1 = Wh + (size_t)(n0 + 16 + r16) * Kp;  // B tile 1: col = n0+16+lane&15
    v8f acc0 = {}, acc1 = {};
    for (int k0 = 0; k0 < Kp; k0 += 32) {
        // A 16x32 f16: lane holds K-run [k0+hi*16..+15]
        v8h a0 = *(const v8h*)(xr + k0 + hi * 16);
        v8h a1 = *(const v8h*)(xr + k0 + hi * 16 + 8);
        v16h A;
#pragma unroll
        for (int j = 0; j < 8; ++j) { A[j] = a0[j]; A[8 + j] = a1[j]; }
        // B 32x16 f16: lane holds column, K = k0 + hi*16 + j  -> contiguous in Wh row
        int kb = k0 + hi * 16;
        v8h b0a = *(const v8h*)(wr0 + kb);
        v8h b0b = *(const v8h*)(wr0 + kb + 8);
        v8h b1a = *(const v8h*)(wr1 + kb);
        v8h b1b = *(const v8h*)(wr1 + kb + 8);
        v16h B0, B1;
#pragma unroll
        for (int j = 0; j < 8; ++j) {
            B0[j] = b0a[j]; B0[8 + j] = b0b[j];
            B1[j] = b1a[j]; B1[8 + j] = b1b[j];
        }
        acc0 = __builtin_amdgcn_wmma_f32_16x16x32_f16(false, A, false, B0, (short)0, acc0, false, false);
        acc1 = __builtin_amdgcn_wmma_f32_16x16x32_f16(false, A, false, B1, (short)0, acc1, false, false);
    }
    float2 sc0 = so[n0 + r16];
    float2 sc1 = so[n0 + 16 + r16];
#pragma unroll
    for (int r = 0; r < 8; ++r) {        // C/D layout: VGPR r -> M = r + hi*8, N = lane&15
        size_t mrow = (size_t)(m0 + r + hi * 8) * Cout;
        float y0 = acc0[r] * sc0.x + sc0.y;
        float y1 = acc1[r] * sc1.x + sc1.y;
        Y[mrow + n0 + r16]      = (_Float16)(y0 > 0.f ? y0 : 0.f);
        Y[mrow + n0 + 16 + r16] = (_Float16)(y1 > 0.f ? y1 : 0.f);
    }
}

// ---------------------------------------------------------------- max over 32 samples
__global__ void maxpool_kernel(const _Float16* __restrict__ Y, _Float16* __restrict__ out,
                               int nctrs, int C) {
    int id = blockIdx.x * blockDim.x + threadIdx.x;
    if (id >= nctrs * C) return;
    int j = id / C, o = id % C;
    const _Float16* p = Y + ((size_t)j * 32) * C + o;
    float m = (float)p[0];
    for (int s = 1; s < 32; ++s) m = fmaxf(m, (float)p[(size_t)s * C]);
    out[id] = (_Float16)m;
}

// ---------------------------------------------------------------- 3-NN + inverse-distance weights
__global__ void knn3_kernel(const float* __restrict__ toxyz, const float* __restrict__ fromxyz,
                            int* __restrict__ idx3, float* __restrict__ w3,
                            int n_to, int n_from, int total) {
    int i = blockIdx.x * blockDim.x + threadIdx.x;
    if (i >= total) return;
    int b = i / n_to;
    const float* fx = fromxyz + (size_t)b * n_from * 3;
    float tx = toxyz[i*3], ty = toxyz[i*3+1], tz = toxyz[i*3+2];
    float d0 = 1e30f, d1 = 1e30f, d2 = 1e30f; int i0 = 0, i1 = 0, i2 = 0;
    for (int j = 0; j < n_from; ++j) {
        float dx = fx[j*3]-tx, dy = fx[j*3+1]-ty, dz = fx[j*3+2]-tz;
        float d = dx*dx + dy*dy + dz*dz;
        if (d < d0)      { d2 = d1; i2 = i1; d1 = d0; i1 = i0; d0 = d; i0 = j; }
        else if (d < d1) { d2 = d1; i2 = i1; d1 = d; i1 = j; }
        else if (d < d2) { d2 = d; i2 = j; }
    }
    float w0 = 1.f/(d0 + 1e-8f), w1 = 1.f/(d1 + 1e-8f), w2 = 1.f/(d2 + 1e-8f);
    float sw = w0 + w1 + w2;
    idx3[i*3] = i0; idx3[i*3+1] = i1; idx3[i*3+2] = i2;
    w3[i*3] = w0/sw; w3[i*3+1] = w1/sw; w3[i*3+2] = w2/sw;
}

// ---------------------------------------------------------------- interpolate + concat -> f16 X
__global__ void interp_concat_kernel(const _Float16* __restrict__ tofeat, int c_to,
                                     const _Float16* __restrict__ fromfeat, int c_from,
                                     const int* __restrict__ idx3, const float* __restrict__ w3,
                                     _Float16* __restrict__ G, int n_to, int n_from, int cin, int total) {
    int id = blockIdx.x * blockDim.x + threadIdx.x;
    if (id >= total) return;
    int i = id / cin, ch = id % cin;
    _Float16 v;
    if (ch < c_to) {
        v = tofeat[(size_t)i * c_to + ch];
    } else {
        int cc = ch - c_to;
        int b = i / n_to;
        const _Float16* ff = fromfeat + (size_t)b * n_from * c_from;
        float a = 0.f;
        for (int k = 0; k < 3; ++k)
            a += w3[i*3+k] * (float)ff[(size_t)idx3[i*3+k] * c_from + cc];
        v = (_Float16)a;
    }
    G[(size_t)i * cin + ch] = v;
}

// ---------------------------------------------------------------- head2 (128->13) + log_softmax
__global__ void head_final_kernel(const _Float16* __restrict__ H, const float* __restrict__ W2,
                                  const float* __restrict__ b2, float* __restrict__ out, int total) {
    int i = blockIdx.x * blockDim.x + threadIdx.x;
    if (i >= total) return;
    const _Float16* h = H + (size_t)i * 128;
    float lg[13];
    for (int o = 0; o < 13; ++o) {
        float s = b2[o];
        const float* wr = W2 + o * 128;
        for (int k = 0; k < 128; ++k) s += (float)h[k] * wr[k];
        lg[o] = s;
    }
    float mx = lg[0];
    for (int o = 1; o < 13; ++o) mx = fmaxf(mx, lg[o]);
    float se = 0.f;
    for (int o = 0; o < 13; ++o) se += expf(lg[o] - mx);
    float lse = mx + logf(se);
    float* po = out + (size_t)i * 13;
    for (int o = 0; o < 13; ++o) po[o] = lg[o] - lse;
}

// ================================================================ host side
struct Lyr  { const float *W, *b, *g, *be, *mn, *vr; };
struct PLyr { const _Float16* Wh; const float2* so; };

static void run_gemm(const _Float16* X, int M, int Kp, const PLyr& P,
                     _Float16* Y, int Cout, hipStream_t st) {
    dim3 grid(M / 64, Cout / 32);
    gemm_bn_relu_kernel<<<grid, 128, 0, st>>>(X, Kp, P.Wh, P.so, Y, Cout);
}

extern "C" void kernel_launch(void* const* d_in, const int* in_sizes, int n_in,
                              void* d_out, int out_size, void* d_ws, size_t ws_size,
                              hipStream_t stream) {
    (void)n_in; (void)out_size; (void)ws_size;

    // ---- parse inputs: disambiguate flatten order via in_sizes ----
    Lyr sa[4][3]; Lyr fp[4][3]; Lyr h1; const float *W2 = nullptr, *b2 = nullptr;
    const float* xin = nullptr;
    const int fp_nlyr[4] = {2, 2, 2, 3};
    int p = 0;
    bool x_first = (in_sizes[0] == BATCH * 4096 * 9);
    if (x_first) { xin = (const float*)d_in[0]; p = 1; }
    bool sa_first = (in_sizes[p] == 32 * 12);   // sa[0][0].W = 384 elems vs fp[0][0].W = 196608
    bool gammaFirst = sa_first;                 // insertion-order dicts store gamma before beta
    auto readLayer = [&](Lyr& L) {
        L.W = (const float*)d_in[p++]; L.b = (const float*)d_in[p++];
        if (gammaFirst) { L.g = (const float*)d_in[p++]; L.be = (const float*)d_in[p++]; }
        else            { L.be = (const float*)d_in[p++]; L.g = (const float*)d_in[p++]; }
        L.mn = (const float*)d_in[p++]; L.vr = (const float*)d_in[p++];
    };
    if (sa_first) {
        for (int i = 0; i < 4; ++i) for (int l = 0; l < 3; ++l) readLayer(sa[i][l]);
        for (int i = 0; i < 4; ++i) for (int l = 0; l < fp_nlyr[i]; ++l) readLayer(fp[i][l]);
        readLayer(h1);
        W2 = (const float*)d_in[p++]; b2 = (const float*)d_in[p++];
    } else {
        for (int i = 0; i < 4; ++i) for (int l = 0; l < fp_nlyr[i]; ++l) readLayer(fp[i][l]);
        readLayer(h1);
        W2 = (const float*)d_in[p++]; b2 = (const float*)d_in[p++];
        for (int i = 0; i < 4; ++i) for (int l = 0; l < 3; ++l) readLayer(sa[i][l]);
    }
    if (!x_first) xin = (const float*)d_in[p++];

    // ---- workspace layout (bump allocator, 256B aligned) ----
    size_t off = 0;
    auto alloc = [&](size_t bytes) -> char* {
        char* r = (char*)d_ws + off;
        off += (bytes + 255) & ~(size_t)255;
        return r;
    };
    const int nl[5] = {4096, 1024, 256, 64, 16};
    const int cL[5] = {9, 64, 128, 256, 512};
    float* xyzL[5];
    for (int l = 0; l < 5; ++l) xyzL[l] = (float*)alloc((size_t)BATCH * nl[l] * 3 * sizeof(float));
    _Float16* fL[5]; fL[0] = nullptr;
    for (int l = 1; l < 5; ++l) fL[l] = (_Float16*)alloc((size_t)BATCH * nl[l] * cL[l] * 2);
    _Float16* G  = (_Float16*)alloc((size_t)8388608 * 2);    // max grouped/interp X (SA0: 262144x32)
    _Float16* Y0 = (_Float16*)alloc((size_t)16777216 * 2);   // max layer out (SA0 L3: 262144x64)
    _Float16* Y1 = (_Float16*)alloc((size_t)16777216 * 2);
    _Float16* fpOut[4];
    const int fp_nto[4]  = {64, 256, 1024, 4096};
    const int fp_cout[4] = {256, 256, 128, 128};
    for (int i = 0; i < 4; ++i) fpOut[i] = (_Float16*)alloc((size_t)BATCH * fp_nto[i] * fp_cout[i] * 2);
    int*   kidx = (int*)alloc((size_t)BATCH * 4096 * 3 * 4);
    float* kw   = (float*)alloc((size_t)BATCH * 4096 * 3 * 4);

    // ---- prepare all layers: f16 weights (padded) + folded (scale, offset) ----
    auto prep = [&](const Lyr& L, int Ktrue, int Kp, int Cout) -> PLyr {
        _Float16* Wh = (_Float16*)alloc((size_t)Cout * Kp * 2);
        float2*   so = (float2*)alloc((size_t)Cout * sizeof(float2));
        int tot = Cout * Kp;
        prep_layer_kernel<<<(tot + 255) / 256, 256, 0, stream>>>(
            L.W, L.b, L.g, L.be, L.mn, L.vr, Wh, so, Ktrue, Kp, Cout);
        PLyr P; P.Wh = Wh; P.so = so; return P;
    };
    const int cinT[4] = {12, 67, 131, 259};
    const int Kps[4]  = {32, 96, 160, 288};
    const int wid[4][3] = {{32,32,64},{64,64,128},{128,128,256},{256,256,512}};
    const int fp_cin[4] = {768, 384, 320, 128};
    const int fp_w[4][3] = {{256,256,0},{256,256,0},{256,128,0},{256,128,128}};
    PLyr Psa[4][3], Pfp[4][3], Ph1;
    for (int l = 0; l < 4; ++l) {
        Psa[l][0] = prep(sa[l][0], cinT[l], Kps[l], wid[l][0]);
        Psa[l][1] = prep(sa[l][1], wid[l][0], wid[l][0], wid[l][1]);
        Psa[l][2] = prep(sa[l][2], wid[l][1], wid[l][1], wid[l][2]);
    }
    for (int i = 0; i < 4; ++i) {
        int k = fp_cin[i];
        for (int l = 0; l < fp_nlyr[i]; ++l) {
            Pfp[i][l] = prep(fp[i][l], k, k, fp_w[i][l]);
            k = fp_w[i][l];
        }
    }
    Ph1 = prep(h1, 128, 128, 128);

    // ---- extract xyz from 9-channel input ----
    {
        int tot = BATCH * 4096 * 3;
        extract_xyz_kernel<<<(tot + 255) / 256, 256, 0, stream>>>(xin, xyzL[0], tot);
    }

    // ---- set-abstraction stages ----
    const float radii[4] = {0.1f, 0.2f, 0.4f, 0.8f};
    for (int l = 0; l < 4; ++l) {
        fps_kernel<<<BATCH, 256, 0, stream>>>(xyzL[l], xyzL[l + 1], nl[l], nl[l + 1]);
        int np = nl[l + 1];
        int ctrs = BATCH * np;
        ball_group_kernel<<<ctrs / 4, 128, 0, stream>>>(
            xyzL[l], xyzL[l + 1],
            (l == 0) ? xin : nullptr, (l == 0) ? nullptr : fL[l], cL[l],
            G, nl[l], np, radii[l] * radii[l], Kps[l]);
        int M = ctrs * 32;
        run_gemm(G,  M, Kps[l],    Psa[l][0], Y0, wid[l][0], stream);
        run_gemm(Y0, M, wid[l][0], Psa[l][1], Y1, wid[l][1], stream);
        run_gemm(Y1, M, wid[l][1], Psa[l][2], Y0, wid[l][2], stream);
        int tot = ctrs * wid[l][2];
        maxpool_kernel<<<(tot + 255) / 256, 256, 0, stream>>>(Y0, fL[l + 1], ctrs, wid[l][2]);
    }

    // ---- feature propagation stages ----
    const int fp_cfrom[4] = {512, 256, 256, 128};
    const int fp_cto[4]   = {256, 128, 64, 0};
    for (int i = 0; i < 4; ++i) {
        int tl = 3 - i;
        int n_to = nl[tl], n_from = nl[tl + 1];
        int tot_to = BATCH * n_to;
        knn3_kernel<<<(tot_to + 255) / 256, 256, 0, stream>>>(
            xyzL[tl], xyzL[tl + 1], kidx, kw, n_to, n_from, tot_to);
        const _Float16* fromf = (i == 0) ? fL[4] : fpOut[i - 1];
        const _Float16* tof   = (i < 3) ? fL[tl] : nullptr;
        int cin = fp_cin[i];
        int tot = tot_to * cin;
        interp_concat_kernel<<<(tot + 255) / 256, 256, 0, stream>>>(
            tof, fp_cto[i], fromf, fp_cfrom[i], kidx, kw, G, n_to, n_from, cin, tot);
        const _Float16* cur = G;
        int curK = cin;
        for (int l2 = 0; l2 < fp_nlyr[i]; ++l2) {
            _Float16* dst = (l2 == fp_nlyr[i] - 1) ? fpOut[i] : ((cur == Y0) ? Y1 : Y0);
            run_gemm(cur, tot_to, curK, Pfp[i][l2], dst, fp_w[i][l2], stream);
            cur = dst; curK = fp_w[i][l2];
        }
    }

    // ---- head: 128->128 (BN+ReLU) then 128->13 + log_softmax ----
    run_gemm(fpOut[3], BATCH * 4096, 128, Ph1, Y0, 128, stream);
    head_final_kernel<<<(BATCH * 4096 + 255) / 256, 256, 0, stream>>>(
        Y0, W2, b2, (float*)d_out, BATCH * 4096);
}
